// Method_77025943487085
// MI455X (gfx1250) — matrix operand order repaired and checked by reading, not compiled, not run on hardware
//
#include <hip/hip_runtime.h>
#include <hip/hip_bf16.h>

typedef __attribute__((ext_vector_type(16))) _Float16 v16h;
typedef __attribute__((ext_vector_type(8)))  _Float16 v8h;
typedef __attribute__((ext_vector_type(8)))  float    v8f;

namespace {
constexpr int kWAY = 5;
constexpr int kNQ  = 4096;
constexpr float kSCALE = 0.27735009811261457f;   // 13^-0.5
}

// ---------------- WMMA helpers (gfx1250 wave32 layouts, ISA 7.12.2) -----------

static __device__ inline v8f wmma16(v16h a, v16h b, v8f c) {
  return __builtin_amdgcn_wmma_f32_16x16x32_f16(false, a, false, b, (short)0, c,
                                                false, false);
}

// Fragment-packed operand: frag f, lane L owns 16 contiguous halfs (32 B).
static __device__ inline v16h load_frag(const _Float16* P, int frag, int lane) {
  return *(const v16h*)(P + ((size_t)(frag * 32 + lane)) * 16);
}

// A fragment from a row-major LDS tile (lda in halfs, 16B-aligned chunks):
// lanes 0-15: row=m0+lane, K k0+0..15 ; lanes 16-31: K k0+16..31.
static __device__ inline v16h load_a_lds(const _Float16* A, int lda, int m0, int k0,
                                         int lane) {
  const _Float16* p = A + (m0 + (lane & 15)) * lda + k0 + ((lane >> 4) << 4);
  v8h lo = *(const v8h*)p;
  v8h hi = *(const v8h*)(p + 8);
  return __builtin_shufflevector(lo, hi, 0, 1, 2, 3, 4, 5, 6, 7,
                                 8, 9, 10, 11, 12, 13, 14, 15);
}

// C/D 16x16 f32: lanes 0-15 rows m0..m0+7, lanes 16-31 rows m0+8..m0+15.
static __device__ inline void store_c(float* C, int ldc, int m0, int n0, v8f c,
                                      int lane) {
  float* p = C + (m0 + ((lane >> 4) << 3)) * ldc + n0 + (lane & 15);
#pragma unroll
  for (int r = 0; r < 8; ++r) p[r * ldc] = c[r];
}

// GEMM with both operands fragment-packed (A frags: mt*KT+kt, B frags: nt*KT+kt)
template <int MT, int NT, int KT>
static __device__ inline void gemm_ff(const _Float16* Af, const _Float16* Bf,
                                      float* C, int ldc, int lane) {
#pragma unroll
  for (int mt = 0; mt < MT; ++mt)
#pragma unroll
    for (int nt = 0; nt < NT; ++nt) {
      v8f acc = {};
#pragma unroll
      for (int kt = 0; kt < KT; ++kt)
        acc = wmma16(load_frag(Af, mt * KT + kt, lane),
                     load_frag(Bf, nt * KT + kt, lane), acc);
      store_c(C, ldc, mt * 16, nt * 16, acc, lane);
    }
}

// GEMM with A in row-major LDS, B fragment-packed in global
template <int MT, int NT, int KT>
static __device__ inline void gemm_lb(const _Float16* A, int lda, const _Float16* Bf,
                                      float* C, int ldc, int lane) {
#pragma unroll
  for (int mt = 0; mt < MT; ++mt)
#pragma unroll
    for (int nt = 0; nt < NT; ++nt) {
      v8f acc = {};
#pragma unroll
      for (int kt = 0; kt < KT; ++kt)
        acc = wmma16(load_a_lds(A, lda, mt * 16, kt * 32, lane),
                     load_frag(Bf, nt * KT + kt, lane), acc);
      store_c(C, ldc, mt * 16, nt * 16, acc, lane);
    }
}

// ---------------- prologue: pack operands into WMMA fragment order ------------

// Support: mean-subtract, L2-norm; emit A-frags (MT=2,KT=2 -> 4 frags/way),
// projection sp, and mean-subtracted spt for the final pooling.
__global__ void k_support(const float* __restrict__ spt,
                          const float* __restrict__ proj_w,
                          const float* __restrict__ proj_b,
                          float* __restrict__ spt_ms, _Float16* __restrict__ sAf,
                          float* __restrict__ sp) {
  const int w = blockIdx.x, t = threadIdx.x;
  const int mt = t >> 4, l15 = t & 15;
  if (t < 25) {
    float vals[64];
    float m = 0.f;
    for (int c = 0; c < 64; ++c) { vals[c] = spt[(w * 64 + c) * 25 + t]; m += vals[c]; }
    m *= (1.f / 64.f);
    float ss = 0.f, pr = 0.f;
    for (int c = 0; c < 64; ++c) {
      float v = vals[c] - m;
      vals[c] = v;
      ss += v * v;
      pr += v * proj_w[c];
      spt_ms[(w * 64 + c) * 25 + t] = v;
    }
    float inv = rsqrtf(ss + 1e-6f);  // FeatureL2Norm
#pragma unroll
    for (int kt = 0; kt < 2; ++kt)
#pragma unroll
      for (int Lh = 0; Lh < 2; ++Lh) {
        const int L = l15 + Lh * 16;
        const int f = mt * 2 + kt;
        const int kb = kt * 32 + Lh * 16;
        v16h a;
#pragma unroll
        for (int i = 0; i < 8; ++i) {
          a[i]     = (_Float16)(vals[kb + i] * inv);
          a[i + 8] = (_Float16)(vals[kb + 8 + i] * inv);
        }
        *(v16h*)(sAf + ((size_t)(w * 4 + f) * 32 + L) * 16) = a;
      }
    sp[w * 25 + t] = pr + proj_b[0];
  } else {  // pad rows 25..31 (mt=1, l15=9..15)
    v16h z = {};
#pragma unroll
    for (int kt = 0; kt < 2; ++kt)
#pragma unroll
      for (int Lh = 0; Lh < 2; ++Lh)
        *(v16h*)(sAf + ((size_t)(w * 4 + mt * 2 + kt) * 32 + l15 + Lh * 16) * 16) = z;
  }
}

// Query: mean-subtract, L2-norm; emit B-frags (NT=2,KT=2 -> 4 frags/query),
// projection tp, and mean-subtracted qry for the final pooling.
__global__ void k_query(const float* __restrict__ qry,
                        const float* __restrict__ proj_w,
                        const float* __restrict__ proj_b,
                        float* __restrict__ qry_ms, _Float16* __restrict__ tBf,
                        float* __restrict__ tp) {
  const int q = blockIdx.x, t = threadIdx.x;
  const size_t base = (size_t)q * 64 * 25;
  const int nt = t >> 4, l15 = t & 15;
  _Float16* dst = tBf + (size_t)q * 4 * 512;
  if (t < 25) {
    float vals[64];
    float m = 0.f;
    for (int c = 0; c < 64; ++c) { vals[c] = qry[base + c * 25 + t]; m += vals[c]; }
    m *= (1.f / 64.f);
    float ss = 0.f, pr = 0.f;
    for (int c = 0; c < 64; ++c) {
      float v = vals[c] - m;
      vals[c] = v;
      ss += v * v;
      pr += v * proj_w[c];
      qry_ms[base + c * 25 + t] = v;
    }
    float inv = rsqrtf(ss + 1e-6f);
#pragma unroll
    for (int kt = 0; kt < 2; ++kt)
#pragma unroll
      for (int Lh = 0; Lh < 2; ++Lh) {
        const int L = l15 + Lh * 16;
        const int f = nt * 2 + kt;
        const int kb = kt * 32 + Lh * 16;   // B frag: 16 consecutive K per lane-half
        v16h b;
#pragma unroll
        for (int i = 0; i < 16; ++i) b[i] = (_Float16)(vals[kb + i] * inv);
        *(v16h*)(dst + ((size_t)(f * 32) + L) * 16) = b;
      }
    tp[q * 25 + t] = pr + proj_b[0];
  } else {  // pad columns 25..31 (nt=1, l15=9..15)
    v16h z = {};
#pragma unroll
    for (int kt = 0; kt < 2; ++kt)
#pragma unroll
      for (int Lh = 0; Lh < 2; ++Lh)
        *(v16h*)(dst + ((size_t)((nt * 2 + kt) * 32) + l15 + Lh * 16) * 16) = z;
  }
}

// Pack a weight matrix W (N_out x K_in, row-major) as B^T fragments of W^T.
static __device__ inline void pack_bfrags(const float* __restrict__ W, int K_in,
                                          int N_out, int NT, int KT,
                                          _Float16* __restrict__ out, int tid,
                                          int nthreads) {
  const int total = NT * KT * 32 * 16;
  for (int idx = tid; idx < total; idx += nthreads) {
    const int i = idx & 15, L = (idx >> 4) & 31, f = idx >> 9;
    const int nt = f / KT, kt = f % KT;
    const int n = nt * 16 + (L & 15);
    const int k = kt * 32 + ((L >> 4) << 4) + i;
    out[idx] = (_Float16)((k < K_in && n < N_out) ? W[n * K_in + k] : 0.f);
  }
}

__global__ void k_pack(const float* __restrict__ pos_x, const float* __restrict__ pos_y,
                       const float* __restrict__ qkv_w, const float* __restrict__ attn_w,
                       const float* __restrict__ fc1_w, const float* __restrict__ fc2_w,
                       const float* __restrict__ dec_w,
                       float* __restrict__ pos, _Float16* __restrict__ Wqkv,
                       _Float16* __restrict__ Wattn, _Float16* __restrict__ Wfc1,
                       _Float16* __restrict__ Wfc2, _Float16* __restrict__ Wdec) {
  const int task = blockIdx.x, tid = threadIdx.x, nt = blockDim.x;
  if (task == 0) {
    for (int i = tid; i < 32 * 32; i += nt) {
      int t = i >> 5, e = i & 31;
      float v = 0.f;
      if (t < 25 && e < 26) {
        int y = t / 5, x = t % 5;
        v = (e < 13) ? pos_x[x * 13 + e] : pos_y[y * 13 + (e - 13)];
      }
      pos[i] = v;
    }
  } else if (task == 1) {
    pack_bfrags(qkv_w, 26, 78, 5, 1, Wqkv, tid, nt);
  } else if (task == 2) {
    pack_bfrags(attn_w, 26, 26, 2, 1, Wattn, tid, nt);
  } else if (task == 3) {
    pack_bfrags(fc1_w, 26, 104, 7, 1, Wfc1, tid, nt);
  } else if (task == 4) {
    pack_bfrags(fc2_w, 104, 26, 2, 4, Wfc2, tid, nt);
  } else if (task == 5) {
    pack_bfrags(dec_w, 26, 25, 2, 1, Wdec, tid, nt);
  }
}

// ---------------- main fused kernel: one wave per (query, way) pair -----------

static __device__ inline void ln_to_f16(const float* __restrict__ xb,
                                        _Float16* __restrict__ hA,
                                        const float* __restrict__ gamma,
                                        const float* __restrict__ beta, int lane) {
  if (lane < 25) {
    const float* row = xb + lane * 32;
    float m = 0.f;
#pragma unroll
    for (int c = 0; c < 26; ++c) m += row[c];
    m *= (1.f / 26.f);
    float v = 0.f;
#pragma unroll
    for (int c = 0; c < 26; ++c) { float d = row[c] - m; v += d * d; }
    float r = rsqrtf(v * (1.f / 26.f) + 1e-6f);
#pragma unroll
    for (int c = 0; c < 26; ++c)
      hA[lane * 32 + c] = (_Float16)((row[c] - m) * r * gamma[c] + beta[c]);
  }
}

__global__ void __launch_bounds__(32) k_main(
    const _Float16* __restrict__ sAf, const _Float16* __restrict__ tBf,
    const float* __restrict__ sp, const float* __restrict__ tp,
    const float* __restrict__ spt_ms, const float* __restrict__ qry_ms,
    const float* __restrict__ pos,
    const _Float16* __restrict__ Wqkv, const _Float16* __restrict__ Wattn,
    const _Float16* __restrict__ Wfc1, const _Float16* __restrict__ Wfc2,
    const _Float16* __restrict__ Wdec,
    const float* __restrict__ ln1_g, const float* __restrict__ ln1_b,
    const float* __restrict__ ln2_g, const float* __restrict__ ln2_b,
    const float* __restrict__ qkv_b, const float* __restrict__ attn_b,
    const float* __restrict__ fc1_b, const float* __restrict__ fc2_b,
    const float* __restrict__ dec_b, float* __restrict__ out) {
  __shared__ float c0[32 * 32];                  // corr, later reused for pq
  __shared__ float c1[32 * 32];                  // pass-2 corr / refined
  __shared__ float xb[32 * 32];                  // x (25x26), later reused for ps
  __shared__ float o32[32 * 32];                 // attention output
  __shared__ float g[32 * 112];                  // GEMM results
  __shared__ __align__(32) _Float16 hA[32 * 32]; // f16 A staging
  __shared__ __align__(32) _Float16 hB[32 * 128];// f16 A staging for fc2
  __shared__ float a_s[32], a_q[32], red[96];

  const int lane = threadIdx.x;
  const int b = blockIdx.x;
  const int q = b / kWAY, w = b % kWAY;

  // zero pads of the f16 staging buffers once; compute loops only touch the
  // valid (row<25, col<26/104) region so pads stay zero for both passes.
  for (int i = lane; i < 32 * 32; i += 32) hA[i] = (_Float16)0.f;
  for (int i = lane; i < 32 * 128; i += 32) hB[i] = (_Float16)0.f;
  __syncthreads();

  // corr(25x25) = s^T(25x64) x t(64x25): both operands frag-packed, 8 WMMAs
  gemm_ff<2, 2, 2>(sAf + (size_t)w * 4 * 512, tBf + (size_t)q * 4 * 512, c0, 32, lane);
  __syncthreads();

  for (int pass = 0; pass < 2; ++pass) {
    // ---- build x: concat(corr-like, proj) + pos, padded to 32x32 ----
    for (int i = lane; i < 32 * 32; i += 32) {
      int t = i >> 5, e = i & 31;
      float v = 0.f;
      if (t < 25 && e < 26) {
        if (pass == 0)
          v = (e < 25) ? c0[e * 32 + t] : tp[q * 25 + t];   // corr^T | tgt
        else
          v = (e < 25) ? c1[t * 32 + e] : sp[w * 25 + t];   // y1^T+corr | src
        v += pos[t * 32 + e];
      }
      xb[i] = v;
    }
    __syncthreads();

    // ---- LN1 -> hA ----
    ln_to_f16(xb, hA, ln1_g, ln1_b, lane);
    __syncthreads();

    // ---- qkv: (25x26)x(26x78) -> 10 WMMAs ----
    gemm_lb<2, 5, 1>(hA, 32, Wqkv, g, 112, lane);
    __syncthreads();
    if (lane < 25) {
      float* gr = g + lane * 112;
#pragma unroll
      for (int c = 0; c < 78; ++c) gr[c] += qkv_b[c];
    }
    __syncthreads();

    // ---- attention: 2 heads, hd=13, T=25 (scalar VALU, lane = query row) ----
    if (lane < 25) {
      const float* gr = g + lane * 112;
#pragma unroll
      for (int h = 0; h < 2; ++h) {
        const int qo = h * 13, ko = 26 + h * 13, vo = 52 + h * 13;
        float sc[25];
        float mx = -1e30f;
#pragma unroll
        for (int j = 0; j < 25; ++j) {
          float s = 0.f;
#pragma unroll
          for (int d = 0; d < 13; ++d) s += gr[qo + d] * g[j * 112 + ko + d];
          s *= kSCALE;
          sc[j] = s;
          mx = fmaxf(mx, s);
        }
        float se = 0.f;
#pragma unroll
        for (int j = 0; j < 25; ++j) { sc[j] = __expf(sc[j] - mx); se += sc[j]; }
        float inv = 1.f / se;
#pragma unroll
        for (int d = 0; d < 13; ++d) {
          float o = 0.f;
#pragma unroll
          for (int j = 0; j < 25; ++j) o += sc[j] * g[j * 112 + vo + d];
          o32[lane * 32 + h * 13 + d] = o * inv;
        }
      }
    }
    __syncthreads();

    // ---- attn out-proj + residual: 4 WMMAs ----
    if (lane < 25) {
#pragma unroll
      for (int c = 0; c < 26; ++c) hA[lane * 32 + c] = (_Float16)o32[lane * 32 + c];
    }
    __syncthreads();
    gemm_lb<2, 2, 1>(hA, 32, Wattn, g, 112, lane);
    __syncthreads();
    if (lane < 25) {
#pragma unroll
      for (int c = 0; c < 26; ++c) xb[lane * 32 + c] += g[lane * 112 + c] + attn_b[c];
    }
    __syncthreads();

    // ---- LN2 -> hA ----
    ln_to_f16(xb, hA, ln2_g, ln2_b, lane);
    __syncthreads();

    // ---- fc1 (25x26 -> 25x104): 14 WMMAs, exact GELU -> hB ----
    gemm_lb<2, 7, 1>(hA, 32, Wfc1, g, 112, lane);
    __syncthreads();
    if (lane < 25) {
#pragma unroll
      for (int c = 0; c < 104; ++c) {
        float v = g[lane * 112 + c] + fc1_b[c];
        v = 0.5f * v * (1.f + erff(v * 0.70710678118654752f));
        hB[lane * 128 + c] = (_Float16)v;
      }
    }
    __syncthreads();

    // ---- fc2 (25x104 -> 25x26): K=128 padded, 16 WMMAs + residual ----
    gemm_lb<2, 2, 4>(hB, 128, Wfc2, g, 112, lane);
    __syncthreads();
    if (lane < 25) {
#pragma unroll
      for (int c = 0; c < 26; ++c) xb[lane * 32 + c] += g[lane * 112 + c] + fc2_b[c];
    }
    __syncthreads();

    // ---- decoder proj (25x26 -> 25x25): 4 WMMAs ----
    if (lane < 25) {
#pragma unroll
      for (int c = 0; c < 26; ++c) hA[lane * 32 + c] = (_Float16)xb[lane * 32 + c];
    }
    __syncthreads();
    gemm_lb<2, 2, 1>(hA, 32, Wdec, g, 112, lane);
    __syncthreads();
    if (lane < 25) {
      if (pass == 0) {  // c1 = y1^T + corr
#pragma unroll
        for (int j = 0; j < 25; ++j)
          c1[lane * 32 + j] = g[j * 112 + lane] + dec_b[lane] + c0[lane * 32 + j];
      } else {          // refined = y2 + corr
#pragma unroll
        for (int j = 0; j < 25; ++j)
          c1[lane * 32 + j] = g[lane * 112 + j] + dec_b[j] + c0[lane * 32 + j];
      }
    }
    __syncthreads();
  }

  // ---- attn_s: column-wise gnorm (ddof=1) + softmax over i, lane = column j ----
  if (lane < 25) {
    float m = 0.f;
#pragma unroll
    for (int i = 0; i < 25; ++i) m += c1[i * 32 + lane];
    m *= (1.f / 25.f);
    float v = 0.f;
#pragma unroll
    for (int i = 0; i < 25; ++i) { float d = c1[i * 32 + lane] - m; v += d * d; }
    float r = rsqrtf(v * (1.f / 24.f) + 1e-5f) * (1.f / 5.f);  // /T_ATTN
    float z[25];
    float mx = -1e30f;
#pragma unroll
    for (int i = 0; i < 25; ++i) { z[i] = (c1[i * 32 + lane] - m) * r; mx = fmaxf(mx, z[i]); }
    float se = 0.f;
#pragma unroll
    for (int i = 0; i < 25; ++i) { z[i] = __expf(z[i] - mx); se += z[i]; }
    float inv = 1.f / se;
#pragma unroll
    for (int i = 0; i < 25; ++i) xb[i * 32 + lane] = z[i] * inv;  // ps[i][j]
  }
  // ---- attn_q: row-wise gnorm + softmax over j, lane = row i ----
  if (lane < 25) {
    float m = 0.f;
#pragma unroll
    for (int j = 0; j < 25; ++j) m += c1[lane * 32 + j];
    m *= (1.f / 25.f);
    float v = 0.f;
#pragma unroll
    for (int j = 0; j < 25; ++j) { float d = c1[lane * 32 + j] - m; v += d * d; }
    float r = rsqrtf(v * (1.f / 24.f) + 1e-5f) * (1.f / 5.f);
    float z[25];
    float mx = -1e30f;
#pragma unroll
    for (int j = 0; j < 25; ++j) { z[j] = (c1[lane * 32 + j] - m) * r; mx = fmaxf(mx, z[j]); }
    float se = 0.f;
#pragma unroll
    for (int j = 0; j < 25; ++j) { z[j] = __expf(z[j] - mx); se += z[j]; }
    float inv = 1.f / se;
#pragma unroll
    for (int j = 0; j < 25; ++j) c0[lane * 32 + j] = z[j] * inv;  // pq[i][j]
  }
  __syncthreads();
  if (lane < 25) {
    float s1 = 0.f, s2 = 0.f;
#pragma unroll
    for (int j = 0; j < 25; ++j) {
      s1 += xb[lane * 32 + j];   // attn_s[i] = sum_j ps[i][j]
      s2 += c0[j * 32 + lane];   // attn_q[j] = sum_i pq[i][j]
    }
    a_s[lane] = s1;
    a_q[lane] = s2;
  }
  __syncthreads();

  // ---- attention-weighted pooling + cosine similarity ----
  float dp = 0.f, ssum = 0.f, qsum = 0.f;
  for (int c = lane; c < 64; c += 32) {
    const float* sm = spt_ms + (w * 64 + c) * 25;
    const float* qm = qry_ms + ((size_t)q * 64 + c) * 25;
    float as = 0.f, aq = 0.f;
#pragma unroll
    for (int t = 0; t < 25; ++t) { as += a_s[t] * sm[t]; aq += a_q[t] * qm[t]; }
    as *= 0.04f;  // mean over 25 spatial positions
    aq *= 0.04f;
    dp += as * aq;
    ssum += as * as;
    qsum += aq * aq;
  }
  red[lane] = dp;
  red[32 + lane] = ssum;
  red[64 + lane] = qsum;
  __syncthreads();
  if (lane == 0) {
    float D = 0.f, A = 0.f, B = 0.f;
    for (int l = 0; l < 32; ++l) { D += red[l]; A += red[32 + l]; B += red[64 + l]; }
    float n1 = fmaxf(sqrtf(A), 1e-8f);
    float n2 = fmaxf(sqrtf(B), 1e-8f);
    out[b] = D / (n1 * n2) * 5.0f;  // / temperature 0.2
  }
}

// ---------------- launcher ----------------------------------------------------

extern "C" void kernel_launch(void* const* d_in, const int* in_sizes, int n_in,
                              void* d_out, int out_size, void* d_ws, size_t ws_size,
                              hipStream_t stream) {
  (void)in_sizes; (void)n_in; (void)out_size; (void)ws_size;
  const float* spt    = (const float*)d_in[0];
  const float* qry    = (const float*)d_in[1];
  const float* proj_w = (const float*)d_in[2];
  const float* proj_b = (const float*)d_in[3];
  const float* pos_x  = (const float*)d_in[4];
  const float* pos_y  = (const float*)d_in[5];
  const float* ln1_g  = (const float*)d_in[6];
  const float* ln1_b  = (const float*)d_in[7];
  const float* qkv_w  = (const float*)d_in[8];
  const float* qkv_b  = (const float*)d_in[9];
  const float* attn_w = (const float*)d_in[10];
  const float* attn_b = (const float*)d_in[11];
  const float* ln2_g  = (const float*)d_in[12];
  const float* ln2_b  = (const float*)d_in[13];
  const float* fc1_w  = (const float*)d_in[14];
  const float* fc1_b  = (const float*)d_in[15];
  const float* fc2_w  = (const float*)d_in[16];
  const float* fc2_b  = (const float*)d_in[17];
  const float* dec_w  = (const float*)d_in[18];
  const float* dec_b  = (const float*)d_in[19];

  char* p = (char*)d_ws;
  auto alloc = [&](size_t bytes) {
    char* r = p;
    p += (bytes + 255) & ~(size_t)255;
    return r;
  };
  float*    spt_ms = (float*)   alloc((size_t)kWAY * 64 * 25 * sizeof(float));
  _Float16* sAf    = (_Float16*)alloc((size_t)kWAY * 4 * 512 * sizeof(_Float16));
  float*    sp     = (float*)   alloc((size_t)kWAY * 25 * sizeof(float));
  float*    qry_ms = (float*)   alloc((size_t)kNQ * 64 * 25 * sizeof(float));
  _Float16* tBf    = (_Float16*)alloc((size_t)kNQ * 4 * 512 * sizeof(_Float16));
  float*    tp     = (float*)   alloc((size_t)kNQ * 25 * sizeof(float));
  float*    pos    = (float*)   alloc(32 * 32 * sizeof(float));
  _Float16* Wqkv   = (_Float16*)alloc(5 * 512 * sizeof(_Float16));
  _Float16* Wattn  = (_Float16*)alloc(2 * 512 * sizeof(_Float16));
  _Float16* Wfc1   = (_Float16*)alloc(7 * 512 * sizeof(_Float16));
  _Float16* Wfc2   = (_Float16*)alloc(8 * 512 * sizeof(_Float16));
  _Float16* Wdec   = (_Float16*)alloc(2 * 512 * sizeof(_Float16));

  k_support<<<kWAY, 32, 0, stream>>>(spt, proj_w, proj_b, spt_ms, sAf, sp);
  k_query<<<kNQ, 32, 0, stream>>>(qry, proj_w, proj_b, qry_ms, tBf, tp);
  k_pack<<<6, 128, 0, stream>>>(pos_x, pos_y, qkv_w, attn_w, fc1_w, fc2_w, dec_w,
                                pos, Wqkv, Wattn, Wfc1, Wfc2, Wdec);
  k_main<<<kNQ * kWAY, 32, 0, stream>>>(sAf, tBf, sp, tp, spt_ms, qry_ms, pos,
                                        Wqkv, Wattn, Wfc1, Wfc2, Wdec,
                                        ln1_g, ln1_b, ln2_g, ln2_b,
                                        qkv_b, attn_b, fc1_b, fc2_b, dec_b,
                                        (float*)d_out);
}
